// QCNNLSTM_62010737819696
// MI455X (gfx1250) — compile-verified, matrix operand order checked by reading, not run on hardware
//
#include <hip/hip_runtime.h>
#include <hip/hip_bf16.h>

// ---------------------------------------------------------------------------
// CDNA5 (gfx1250) implementation of the quantized CNN-LSTM-MLP forward.
// GEMM-heavy stages (LSTM gates, per-head MLP) use v_wmma_f32_16x16x32_f16
// with 16x64 register blocking per wave (A fragment reused across 4 col tiles
// to cut L2 activation traffic 4x). Dynamic fake-quant scales / BN stats use
// multi-pass atomic reductions in workspace.
// ---------------------------------------------------------------------------

typedef __attribute__((ext_vector_type(16))) _Float16 v16h;
typedef __attribute__((ext_vector_type(8)))  _Float16 v8h;
typedef __attribute__((ext_vector_type(8)))  float    v8f;

static constexpr int B_  = 8192;
static constexpr int L_  = 32;
static constexpr int C1_ = 64;
static constexpr int C2_ = 128;
static constexpr int H_  = 128;
static constexpr int G4_ = 512;   // 4*H
static constexpr int T_  = 16;
static constexpr int M_  = 256;
static constexpr int NH_ = 4;

// ---- workspace arena (byte offsets) ---------------------------------------
static constexpr size_t OFF_SC   = 0;                    // 256 uints: absmax slots
static constexpr size_t OFF_BN   = 4u * 1024;            // BN stat floats (~26KB)
static constexpr size_t OFF_W2Q  = 40u * 1024;           // conv2 quantized weights f32 (96KB)
static constexpr size_t OFF_WQ16 = 256u * 1024;          // f16 quantized weights (~1.9MB)
static constexpr size_t OFF_A    = 4u * 1024 * 1024;     // region A (80MB): y1/bn1out, seq f32, hs0 f16
static constexpr size_t OFF_B    = 84u * 1024 * 1024;    // region B (136MB): y2 f32 / MLP y / z bufs
static constexpr size_t OFF_GT   = OFF_B + 80u * 1024 * 1024;  // lstm gates f32 (16.8MB)
static constexpr size_t OFF_Z1   = OFF_B + 40u * 1024 * 1024;  // MLP z f16
static constexpr size_t OFF_Z2   = OFF_B + 60u * 1024 * 1024;
static constexpr size_t OFF_C    = 220u * 1024 * 1024;   // region C: seqq f16, h/c state
static constexpr size_t OFF_HST  = OFF_C + 34u * 1024 * 1024;  // h state f16 (2MB)
static constexpr size_t OFF_CST  = OFF_C + 36u * 1024 * 1024;  // c state f32 (4MB)

// f16 weight sub-offsets (in halfs)
static constexpr size_t WQ_WIH0 = 0, WQ_WHH0 = 65536, WQ_WIH1 = 131072, WQ_WHH1 = 196608;
static constexpr size_t WQ_HW1 = 262144, WQ_HW2 = 393216, WQ_HW3 = 655360, WQ_HWO = 917504;

// scale slot indices
// 0:x 1:w1 2:w2 3:y1 4:bn1out 5:y2 6:seq 7..10:wih0,whh0,wih1,whh1
// 12..27: hW1..hWo per-head   32..63: gates[layer*16+t]  64..95: c[layer*16+t]
// 96..99:mlp y1  100..103:bn1o  104..107:y2  108..111:bn2o  112..115:y3  116..119:bn3o

// ---- helpers --------------------------------------------------------------
__device__ __forceinline__ float fq_val(float x, float s, float qmax, float qmin) {
  float q = rintf(x / s);
  q = fminf(fmaxf(q, qmin), qmax);
  return q * s;
}
__device__ __forceinline__ float scale_of(unsigned u, float qmax) {
  return fmaxf(__uint_as_float(u) / qmax, 1e-8f);
}
__device__ __forceinline__ unsigned fabs_bits(float x) { return __float_as_uint(fabsf(x)); }

// blockDim must be 256 for all callers
__device__ __forceinline__ void block_max_atomic(unsigned v, unsigned* dst) {
  __shared__ unsigned red[256];
  red[threadIdx.x] = v;
  __syncthreads();
  for (int s = 128; s > 0; s >>= 1) {
    if ((int)threadIdx.x < s) red[threadIdx.x] = max(red[threadIdx.x], red[threadIdx.x + s]);
    __syncthreads();
  }
  if (threadIdx.x == 0) atomicMax(dst, red[0]);
}

// 16x32 f16 fragment per wave; base row-major with leading dim ld (halfs).
// lane = m (rows 0..15), half selects k-subgroup per ISA layout:
//   element e -> k = 16*(e>>3) + 8*half + (e&7)
__device__ __forceinline__ v16h load_frag16x32(const _Float16* __restrict__ base,
                                               int ld, int row0, int k0) {
  int lane = threadIdx.x & 31;
  int r = row0 + (lane & 15);
  int half = lane >> 4;
  const _Float16* p = base + (size_t)r * ld + k0 + 8 * half;
  v8h lo = *(const v8h*)(p);
  v8h hi = *(const v8h*)(p + 16);
  v16h f;
#pragma unroll
  for (int i = 0; i < 8; ++i) { f[i] = lo[i]; f[i + 8] = hi[i]; }
  return f;
}

__device__ __forceinline__ v8f wmma_f16(v16h a, v16h b, v8f c) {
  return __builtin_amdgcn_wmma_f32_16x16x32_f16(false, a, false, b, (short)0, c, false, false);
}

// ---- generic small kernels ------------------------------------------------
__global__ void k_zero(unsigned* p, int n) {
  int i = blockIdx.x * blockDim.x + threadIdx.x;
  if (i < n) p[i] = 0u;
}

__global__ void k_absmax(const float* __restrict__ p, int n, unsigned* dst) {
  unsigned m = 0;
  for (int i = blockIdx.x * blockDim.x + threadIdx.x; i < n; i += gridDim.x * blockDim.x)
    m = max(m, fabs_bits(p[i]));
  block_max_atomic(m, dst);
}

__global__ void k_absmax_head(const float* __restrict__ p, int per, unsigned* dst) {
  int h = blockIdx.y;
  const float* q = p + (size_t)h * per;
  unsigned m = 0;
  for (int i = blockIdx.x * blockDim.x + threadIdx.x; i < per; i += gridDim.x * blockDim.x)
    m = max(m, fabs_bits(q[i]));
  block_max_atomic(m, dst + h);
}

__global__ void k_quant_f32(const float* __restrict__ src, float* __restrict__ dst, int n,
                            const unsigned* sm, float qmax, float qmin) {
  int i = blockIdx.x * blockDim.x + threadIdx.x;
  if (i >= n) return;
  float s = scale_of(*sm, qmax);
  dst[i] = fq_val(src[i], s, qmax, qmin);
}

__global__ void k_quant_f16(const float* __restrict__ src, _Float16* __restrict__ dst, int n,
                            const unsigned* sm, float qmax, float qmin) {
  int i = blockIdx.x * blockDim.x + threadIdx.x;
  if (i >= n) return;
  float s = scale_of(*sm, qmax);
  dst[i] = (_Float16)fq_val(src[i], s, qmax, qmin);
}

__global__ void k_quant_f16_head(const float* __restrict__ src, _Float16* __restrict__ dst,
                                 int per, const unsigned* sm, float qmax, float qmin) {
  int h = blockIdx.y;
  float s = scale_of(sm[h], qmax);
  for (int i = blockIdx.x * blockDim.x + threadIdx.x; i < per; i += gridDim.x * blockDim.x) {
    size_t a = (size_t)h * per + i;
    dst[a] = (_Float16)fq_val(src[a], s, qmax, qmin);
  }
}

// ---- conv / BN stages -----------------------------------------------------
__global__ void k_conv1(const float* __restrict__ x, const float* __restrict__ w,
                        const float* __restrict__ bias, float* __restrict__ y,
                        const unsigned* sc, unsigned* ymax) {
  int idx = blockIdx.x * blockDim.x + threadIdx.x;   // b*2048 + c*32 + l
  unsigned m = 0;
  if (idx < B_ * C1_ * L_) {
    int l = idx & 31, c = (idx >> 5) & 63, b = idx >> 11;
    float sx = scale_of(sc[0], 127.f), sw = scale_of(sc[1], 127.f);
    float acc = bias[c];
#pragma unroll
    for (int k = 0; k < 3; ++k) {
      int ll = l + k - 1;
      if (0 <= ll && ll < L_)
        acc += fq_val(x[b * L_ + ll], sx, 127.f, -128.f) *
               fq_val(w[c * 3 + k], sw, 127.f, -128.f);
    }
    acc = fmaxf(acc, 0.f);
    y[idx] = acc;
    m = fabs_bits(acc);
  }
  block_max_atomic(m, ymax);
}

// quantize conv activations (6-bit) in place + per-channel sum/sumsq
__global__ void k_qstats(float* __restrict__ y, int C, int N, const unsigned* sm,
                         float* sums, float* sumsq, float qmax, float qmin) {
  int c = blockIdx.x;
  float s = scale_of(*sm, qmax);
  float ls = 0.f, lq = 0.f;
  int CL = C * L_;
  for (int i = threadIdx.x; i < N; i += blockDim.x) {
    int b = i >> 5, l = i & 31;
    size_t a = (size_t)b * CL + (size_t)c * L_ + l;
    float q = fq_val(y[a], s, qmax, qmin);
    y[a] = q;
    ls += q; lq += q * q;
  }
  __shared__ float r1[256], r2[256];
  r1[threadIdx.x] = ls; r2[threadIdx.x] = lq;
  __syncthreads();
  for (int t = 128; t > 0; t >>= 1) {
    if ((int)threadIdx.x < t) { r1[threadIdx.x] += r1[threadIdx.x + t]; r2[threadIdx.x] += r2[threadIdx.x + t]; }
    __syncthreads();
  }
  if (threadIdx.x == 0) { sums[c] = r1[0]; sumsq[c] = r2[0]; }
}

__global__ void k_bn_apply(float* __restrict__ y, int C, const float* sums, const float* sumsq,
                           float invN, const float* g, const float* bta, unsigned* omax, int total) {
  int idx = blockIdx.x * blockDim.x + threadIdx.x;
  unsigned mm = 0;
  if (idx < total) {
    int c = (idx / L_) % C;
    float m = sums[c] * invN, v = sumsq[c] * invN - m * m;
    float o = (y[idx] - m) * rsqrtf(v + 1e-5f) * g[c] + bta[c];
    y[idx] = o;
    mm = fabs_bits(o);
  }
  block_max_atomic(mm, omax);
}

__global__ void k_conv2(const float* __restrict__ hb, const float* __restrict__ w2q,
                        const float* __restrict__ bias, float* __restrict__ y2,
                        const unsigned* sc_in, unsigned* ymax) {
  __shared__ float tile[C1_ * L_];   // 8KB input tile for one batch row
  int b = blockIdx.x;
  float si = scale_of(*sc_in, 127.f);
  for (int i = threadIdx.x; i < C1_ * L_; i += blockDim.x)
    tile[i] = fq_val(hb[(size_t)b * C1_ * L_ + i], si, 127.f, -128.f);
  __syncthreads();
  unsigned m = 0;
  for (int o = threadIdx.x; o < C2_ * L_; o += blockDim.x) {
    int co = o >> 5, l = o & 31;
    float acc = bias[co];
    const float* wr = w2q + co * C1_ * 3;
    for (int ci = 0; ci < C1_; ++ci) {
#pragma unroll
      for (int k = 0; k < 3; ++k) {
        int ll = l + k - 1;
        if (0 <= ll && ll < L_) acc += tile[ci * L_ + ll] * wr[ci * 3 + k];
      }
    }
    acc = fmaxf(acc, 0.f);
    y2[(size_t)b * C2_ * L_ + o] = acc;
    m = max(m, fabs_bits(acc));
  }
  block_max_atomic(m, ymax);
}

// bn2 + AdaptiveAvgPool1d(16) -> seq f32 [T][B][C2]
__global__ void k_bn_pool(const float* __restrict__ y2, const float* sums, const float* sumsq,
                          float invN, const float* g, const float* bta,
                          float* __restrict__ seq, unsigned* smax) {
  int idx = blockIdx.x * blockDim.x + threadIdx.x;
  unsigned mm = 0;
  if (idx < T_ * B_ * C2_) {
    int c = idx % C2_;
    int b = (idx / C2_) % B_;
    int t = idx / (C2_ * B_);
    float m = sums[c] * invN, v = sumsq[c] * invN - m * m;
    float rs = rsqrtf(v + 1e-5f) * g[c];
    size_t base = (size_t)b * C2_ * L_ + (size_t)c * L_ + 2 * t;
    float o = 0.5f * (((y2[base] - m) * rs + bta[c]) + ((y2[base + 1] - m) * rs + bta[c]));
    seq[idx] = o;
    mm = fabs_bits(o);
  }
  block_max_atomic(mm, smax);
}

// ---- LSTM -----------------------------------------------------------------
__global__ void k_init_hc(_Float16* __restrict__ h, float* __restrict__ c, int n) {
  int i = blockIdx.x * blockDim.x + threadIdx.x;
  if (i < n) { h[i] = (_Float16)0.f; c[i] = 0.f; }
}

// gates[B][512] = x@wih^T + h@whh^T + (bih+bhh)
// Each wave: 16 rows x 64 cols (4 WMMA col tiles) -> A fragments loaded once
// per K-chunk and reused 4x. 4 waves/block, grid (B/16, G4/256).
__global__ __launch_bounds__(128)
void k_lstm_gates(const _Float16* __restrict__ x, const _Float16* __restrict__ h,
                  const _Float16* __restrict__ wih, const _Float16* __restrict__ whh,
                  const float* __restrict__ bih, const float* __restrict__ bhh,
                  float* __restrict__ gates, unsigned* gmax) {
  int wave = threadIdx.x >> 5, lane = threadIdx.x & 31;
  int row0 = blockIdx.x * 16;
  int col0 = (blockIdx.y * 4 + wave) * 64;
  v8f acc[4] = {};
#pragma unroll
  for (int k0 = 0; k0 < H_; k0 += 32) {
    v16h a  = load_frag16x32(x, H_, row0, k0);
    v16h a2 = load_frag16x32(h, H_, row0, k0);
#pragma unroll
    for (int j = 0; j < 4; ++j) {
      v16h bw = load_frag16x32(wih, H_, col0 + 16 * j, k0);
      acc[j] = wmma_f16(a, bw, acc[j]);
      v16h b2 = load_frag16x32(whh, H_, col0 + 16 * j, k0);
      acc[j] = wmma_f16(a2, b2, acc[j]);
    }
  }
  int n = lane & 15, half = lane >> 4;
  unsigned mm = 0;
#pragma unroll
  for (int j = 0; j < 4; ++j) {
    int cc = col0 + 16 * j + n;
    float bias = bih[cc] + bhh[cc];
#pragma unroll
    for (int v = 0; v < 8; ++v) {
      int mrow = v + 8 * half;
      float val = acc[j][v] + bias;
      gates[(size_t)(row0 + mrow) * G4_ + cc] = val;
      mm = max(mm, fabs_bits(val));
    }
  }
  atomicMax(gmax, mm);
}

__global__ void k_lstm_point(const float* __restrict__ gates, float* __restrict__ c,
                             _Float16* __restrict__ hst, _Float16* __restrict__ hs_out,
                             unsigned* sc, int gslot, int cprev, int cslot) {
  int idx = blockIdx.x * blockDim.x + threadIdx.x;
  unsigned mm = 0;
  if (idx < B_ * H_) {
    int b = idx >> 7, j = idx & 127;
    float sg = scale_of(sc[gslot], 31.f);
    const float* gr = gates + (size_t)b * G4_;
    float gi = fq_val(gr[j],          sg, 31.f, -32.f);
    float gf = fq_val(gr[H_ + j],     sg, 31.f, -32.f);
    float gg = fq_val(gr[2 * H_ + j], sg, 31.f, -32.f);
    float go = fq_val(gr[3 * H_ + j], sg, 31.f, -32.f);
    const float s6 = 1.f / 31.f, s8 = 1.f / 127.f;   // analytic scales for bounded acts
    float iv = fq_val(1.f / (1.f + __expf(-gi)), s6, 31.f, -32.f);
    float fv = fq_val(1.f / (1.f + __expf(-gf)), s6, 31.f, -32.f);
    float gv = fq_val(tanhf(gg),                 s6, 31.f, -32.f);
    float ov = fq_val(1.f / (1.f + __expf(-go)), s6, 31.f, -32.f);
    float craw = fv * c[idx] + iv * gv;
    float scc = (cprev < 0) ? s6 : fmaxf(scale_of(sc[cprev], 31.f), s6);
    float cq = fq_val(craw, scc, 31.f, -32.f);
    c[idx] = cq;
    float tv = fq_val(tanhf(cq), s6, 31.f, -32.f);
    float hv = fq_val(ov * tv, s8, 127.f, -128.f);
    hst[idx] = (_Float16)hv;
    if (hs_out) hs_out[idx] = (_Float16)hv;
    mm = fabs_bits(craw);
  }
  block_max_atomic(mm, sc + cslot);
}

// ---- per-head MLP ---------------------------------------------------------
// Y[h][b][o] = relu(sum_i A[(h)][b][i] * W[h][o][i] + bias[h][o])
// Each wave: 16 rows x 64 cols; 4 waves/block cover all M=256 cols (grid.y=1).
__global__ __launch_bounds__(128)
void k_mlp_gemm(const _Float16* __restrict__ A, size_t a_head_stride,
                const _Float16* __restrict__ W, const float* __restrict__ bias,
                float* __restrict__ Y, unsigned* hmax, int K) {
  int hh = blockIdx.z;
  int wave = threadIdx.x >> 5, lane = threadIdx.x & 31;
  int row0 = blockIdx.x * 16;
  int col0 = (blockIdx.y * 4 + wave) * 64;
  const _Float16* Ah = A + a_head_stride * hh;
  const _Float16* Wh = W + (size_t)hh * M_ * K;
  v8f acc[4] = {};
  for (int k0 = 0; k0 < K; k0 += 32) {
    v16h a = load_frag16x32(Ah, K, row0, k0);
#pragma unroll
    for (int j = 0; j < 4; ++j) {
      v16h b = load_frag16x32(Wh, K, col0 + 16 * j, k0);
      acc[j] = wmma_f16(a, b, acc[j]);
    }
  }
  int n = lane & 15, half = lane >> 4;
  unsigned mm = 0;
#pragma unroll
  for (int j = 0; j < 4; ++j) {
    int cc = col0 + 16 * j + n;
    float bv = bias[hh * M_ + cc];
#pragma unroll
    for (int v = 0; v < 8; ++v) {
      int mrow = v + 8 * half;
      float val = fmaxf(acc[j][v] + bv, 0.f);
      Y[((size_t)hh * B_ + row0 + mrow) * M_ + cc] = val;
      mm = max(mm, fabs_bits(val));
    }
  }
  atomicMax(hmax + hh, mm);
}

// quantize (6-bit, per-head) in place + per-(h,o) sum/sumsq; one block per (h,o)
__global__ void k_mlp_qstats(float* __restrict__ Y, const unsigned* hmax,
                             float* sums, float* sumsq) {
  int ho = blockIdx.x;
  int hh = ho >> 8, o = ho & 255;
  float s = scale_of(hmax[hh], 31.f);
  float ls = 0.f, lq = 0.f;
  for (int b = threadIdx.x; b < B_; b += blockDim.x) {
    size_t a = ((size_t)hh * B_ + b) * M_ + o;
    float q = fq_val(Y[a], s, 31.f, -32.f);
    Y[a] = q;
    ls += q; lq += q * q;
  }
  __shared__ float r1[256], r2[256];
  r1[threadIdx.x] = ls; r2[threadIdx.x] = lq;
  __syncthreads();
  for (int t = 128; t > 0; t >>= 1) {
    if ((int)threadIdx.x < t) { r1[threadIdx.x] += r1[threadIdx.x + t]; r2[threadIdx.x] += r2[threadIdx.x + t]; }
    __syncthreads();
  }
  if (threadIdx.x == 0) { sums[ho] = r1[0]; sumsq[ho] = r2[0]; }
}

__global__ void k_mlp_bn(float* __restrict__ Y, const float* sums, const float* sumsq,
                         const float* g, const float* bt, unsigned* omax) {
  int idx = blockIdx.x * blockDim.x + threadIdx.x;   // h*B*M + b*M + o
  unsigned mm = 0;
  int hh = (int)(((size_t)blockIdx.x * blockDim.x) / ((size_t)B_ * M_));  // uniform per block
  if (idx < NH_ * B_ * M_) {
    int o = idx % M_;
    int ho = hh * M_ + o;
    const float invN = 1.f / (float)B_;
    float m = sums[ho] * invN, v = sumsq[ho] * invN - m * m;
    float out = (Y[idx] - m) * rsqrtf(v + 1e-5f) * g[ho] + bt[ho];
    Y[idx] = out;
    mm = fabs_bits(out);
  }
  block_max_atomic(mm, omax + hh);
}

__global__ void k_out(const _Float16* __restrict__ z, const _Float16* __restrict__ wo,
                      const float* __restrict__ bo, float* __restrict__ out) {
  int idx = blockIdx.x * blockDim.x + threadIdx.x;   // b*NH + h
  if (idx >= B_ * NH_) return;
  int hh = idx & 3, b = idx >> 2;
  float acc = bo[hh];
  const _Float16* zr = z + ((size_t)hh * B_ + b) * M_;
  const _Float16* wr = wo + hh * M_;
  for (int i = 0; i < M_; ++i) acc += (float)zr[i] * (float)wr[i];
  out[idx] = acc;
}

// ---------------------------------------------------------------------------
extern "C" void kernel_launch(void* const* d_in, const int* in_sizes, int n_in,
                              void* d_out, int out_size, void* d_ws, size_t ws_size,
                              hipStream_t stream) {
  (void)in_sizes; (void)n_in; (void)out_size; (void)ws_size;

  const float* x       = (const float*)d_in[0];
  const float* conv1_w = (const float*)d_in[1];
  const float* conv1_b = (const float*)d_in[2];
  const float* bn1_g   = (const float*)d_in[3];
  const float* bn1_b   = (const float*)d_in[4];
  const float* conv2_w = (const float*)d_in[5];
  const float* conv2_b = (const float*)d_in[6];
  const float* bn2_g   = (const float*)d_in[7];
  const float* bn2_b   = (const float*)d_in[8];
  const float* wih0 = (const float*)d_in[9],  *whh0 = (const float*)d_in[10];
  const float* bih0 = (const float*)d_in[11], *bhh0 = (const float*)d_in[12];
  const float* wih1 = (const float*)d_in[13], *whh1 = (const float*)d_in[14];
  const float* bih1 = (const float*)d_in[15], *bhh1 = (const float*)d_in[16];
  const float* hW1 = (const float*)d_in[17], *hb1 = (const float*)d_in[18];
  const float* hg1 = (const float*)d_in[19], *hbt1 = (const float*)d_in[20];
  const float* hW2 = (const float*)d_in[21], *hb2 = (const float*)d_in[22];
  const float* hg2 = (const float*)d_in[23], *hbt2 = (const float*)d_in[24];
  const float* hW3 = (const float*)d_in[25], *hb3 = (const float*)d_in[26];
  const float* hg3 = (const float*)d_in[27], *hbt3 = (const float*)d_in[28];
  const float* hWo = (const float*)d_in[29], *hbo = (const float*)d_in[30];

  char* ws = (char*)d_ws;
  unsigned* sc   = (unsigned*)(ws + OFF_SC);
  float* bnst    = (float*)(ws + OFF_BN);
  float* w2q     = (float*)(ws + OFF_W2Q);
  _Float16* wq   = (_Float16*)(ws + OFF_WQ16);
  float* bufA    = (float*)(ws + OFF_A);       // y1 / bn1out, later seq f32
  float* y2      = (float*)(ws + OFF_B);
  float* gates   = (float*)(ws + OFF_GT);
  _Float16* seqq = (_Float16*)(ws + OFF_C);
  _Float16* hst  = (_Float16*)(ws + OFF_HST);
  float* cst     = (float*)(ws + OFF_CST);
  _Float16* hs0  = (_Float16*)(ws + OFF_A);    // reuse A after seq consumed
  float* ybuf    = (float*)(ws + OFF_B);       // MLP activations (reuse y2 space)
  _Float16* z1   = (_Float16*)(ws + OFF_Z1);
  _Float16* z2   = (_Float16*)(ws + OFF_Z2);

  // 0) zero scale slots + BN stats region (40KB)
  k_zero<<<40, 256, 0, stream>>>((unsigned*)ws, 10240);

  // 1) static absmax reductions
  k_absmax<<<128, 256, 0, stream>>>(x, B_ * L_, sc + 0);
  k_absmax<<<1,   256, 0, stream>>>(conv1_w, 192, sc + 1);
  k_absmax<<<8,   256, 0, stream>>>(conv2_w, 24576, sc + 2);
  k_absmax<<<32,  256, 0, stream>>>(wih0, G4_ * H_, sc + 7);
  k_absmax<<<32,  256, 0, stream>>>(whh0, G4_ * H_, sc + 8);
  k_absmax<<<32,  256, 0, stream>>>(wih1, G4_ * H_, sc + 9);
  k_absmax<<<32,  256, 0, stream>>>(whh1, G4_ * H_, sc + 10);
  k_absmax_head<<<dim3(8,  NH_), 256, 0, stream>>>(hW1, M_ * H_, sc + 12);
  k_absmax_head<<<dim3(16, NH_), 256, 0, stream>>>(hW2, M_ * M_, sc + 16);
  k_absmax_head<<<dim3(16, NH_), 256, 0, stream>>>(hW3, M_ * M_, sc + 20);
  k_absmax_head<<<dim3(1,  NH_), 256, 0, stream>>>(hWo, M_, sc + 24);

  // 2) quantize weights (f16 for WMMA stages, f32 for conv2)
  k_quant_f32<<<96, 256, 0, stream>>>(conv2_w, w2q, 24576, sc + 2, 127.f, -128.f);
  k_quant_f16<<<256, 256, 0, stream>>>(wih0, wq + WQ_WIH0, G4_ * H_, sc + 7,  127.f, -128.f);
  k_quant_f16<<<256, 256, 0, stream>>>(whh0, wq + WQ_WHH0, G4_ * H_, sc + 8,  127.f, -128.f);
  k_quant_f16<<<256, 256, 0, stream>>>(wih1, wq + WQ_WIH1, G4_ * H_, sc + 9,  127.f, -128.f);
  k_quant_f16<<<256, 256, 0, stream>>>(whh1, wq + WQ_WHH1, G4_ * H_, sc + 10, 127.f, -128.f);
  k_quant_f16_head<<<dim3(128, NH_), 256, 0, stream>>>(hW1, wq + WQ_HW1, M_ * H_, sc + 12, 127.f, -128.f);
  k_quant_f16_head<<<dim3(256, NH_), 256, 0, stream>>>(hW2, wq + WQ_HW2, M_ * M_, sc + 16, 127.f, -128.f);
  k_quant_f16_head<<<dim3(256, NH_), 256, 0, stream>>>(hW3, wq + WQ_HW3, M_ * M_, sc + 20, 127.f, -128.f);
  k_quant_f16_head<<<dim3(1,   NH_), 256, 0, stream>>>(hWo, wq + WQ_HWO, M_, sc + 24, 127.f, -128.f);

  // 3) conv1 + relu (+absmax)
  k_conv1<<<65536, 256, 0, stream>>>(x, conv1_w, conv1_b, bufA, sc, sc + 3);
  // 4) 6-bit quant + per-channel BN stats
  k_qstats<<<C1_, 256, 0, stream>>>(bufA, C1_, B_ * L_, sc + 3, bnst + 0, bnst + 64, 31.f, -32.f);
  // 5) BN1 apply (+absmax for conv2 input quant)
  k_bn_apply<<<65536, 256, 0, stream>>>(bufA, C1_, bnst + 0, bnst + 64, 1.f / (B_ * L_),
                                        bn1_g, bn1_b, sc + 4, B_ * C1_ * L_);
  // 6) conv2 + relu via LDS tile
  k_conv2<<<B_, 256, 0, stream>>>(bufA, w2q, conv2_b, y2, sc + 4, sc + 5);
  // 7) quant + BN2 stats
  k_qstats<<<C2_, 256, 0, stream>>>(y2, C2_, B_ * L_, sc + 5, bnst + 128, bnst + 256, 31.f, -32.f);
  // 8) BN2 + avgpool -> seq f32 (region A) (+absmax)
  k_bn_pool<<<65536, 256, 0, stream>>>(y2, bnst + 128, bnst + 256, 1.f / (B_ * L_),
                                       bn2_g, bn2_b, bufA, sc + 6);
  // 9) seq -> f16 (io quant, 8-bit dynamic)
  k_quant_f16<<<65536, 256, 0, stream>>>(bufA, seqq, T_ * B_ * H_, sc + 6, 127.f, -128.f);

  // 10) two LSTM layers, kernel-per-step (WMMA gates GEMM + pointwise)
  for (int layer = 0; layer < 2; ++layer) {
    k_init_hc<<<4096, 256, 0, stream>>>(hst, cst, B_ * H_);
    const _Float16* wihq = wq + (layer ? WQ_WIH1 : WQ_WIH0);
    const _Float16* whhq = wq + (layer ? WQ_WHH1 : WQ_WHH0);
    const float* bih = layer ? bih1 : bih0;
    const float* bhh = layer ? bhh1 : bhh0;
    for (int t = 0; t < T_; ++t) {
      const _Float16* xin = (layer == 0 ? seqq : hs0) + (size_t)t * B_ * H_;
      int gslot = 32 + layer * 16 + t;
      int cslot = 64 + layer * 16 + t;
      k_lstm_gates<<<dim3(B_ / 16, G4_ / 256), 128, 0, stream>>>(
          xin, hst, wihq, whhq, bih, bhh, gates, sc + gslot);
      _Float16* hsout = (layer == 0) ? hs0 + (size_t)t * B_ * H_ : nullptr;
      k_lstm_point<<<4096, 256, 0, stream>>>(gates, cst, hst, hsout, sc, gslot,
                                             t ? cslot - 1 : -1, cslot);
    }
  }

  // 11) per-head MLP blocks (WMMA) with BN-over-batch and dynamic quant
  const dim3 gmlp(B_ / 16, 1, NH_);   // 4 waves/block cover all 256 cols
  // block 1: input = last hidden state (shared across heads), K=128
  k_mlp_gemm<<<gmlp, 128, 0, stream>>>(hst, (size_t)0, wq + WQ_HW1, hb1, ybuf, sc + 96, H_);
  k_mlp_qstats<<<NH_ * M_, 256, 0, stream>>>(ybuf, sc + 96, bnst + 384, bnst + 1408);
  k_mlp_bn<<<(NH_ * B_ * M_) / 256, 256, 0, stream>>>(ybuf, bnst + 384, bnst + 1408, hg1, hbt1, sc + 100);
  k_quant_f16_head<<<dim3(2048, NH_), 256, 0, stream>>>(ybuf, z1, B_ * M_, sc + 100, 127.f, -128.f);
  // block 2: K=256, per-head input
  k_mlp_gemm<<<gmlp, 128, 0, stream>>>(z1, (size_t)B_ * M_, wq + WQ_HW2, hb2, ybuf, sc + 104, M_);
  k_mlp_qstats<<<NH_ * M_, 256, 0, stream>>>(ybuf, sc + 104, bnst + 2432, bnst + 3456);
  k_mlp_bn<<<(NH_ * B_ * M_) / 256, 256, 0, stream>>>(ybuf, bnst + 2432, bnst + 3456, hg2, hbt2, sc + 108);
  k_quant_f16_head<<<dim3(2048, NH_), 256, 0, stream>>>(ybuf, z2, B_ * M_, sc + 108, 127.f, -128.f);
  // block 3
  k_mlp_gemm<<<gmlp, 128, 0, stream>>>(z2, (size_t)B_ * M_, wq + WQ_HW3, hb3, ybuf, sc + 112, M_);
  k_mlp_qstats<<<NH_ * M_, 256, 0, stream>>>(ybuf, sc + 112, bnst + 4480, bnst + 5504);
  k_mlp_bn<<<(NH_ * B_ * M_) / 256, 256, 0, stream>>>(ybuf, bnst + 4480, bnst + 5504, hg3, hbt3, sc + 116);
  k_quant_f16_head<<<dim3(2048, NH_), 256, 0, stream>>>(ybuf, z1, B_ * M_, sc + 116, 127.f, -128.f);

  // 12) output heads -> d_out [B, NH] f32
  k_out<<<(B_ * NH_) / 256, 256, 0, stream>>>(z1, wq + WQ_HWO, hbo, (float*)d_out);
}
